// Model_82583631167869
// MI455X (gfx1250) — matrix operand order repaired and checked by reading, not compiled, not run on hardware
//
#include <hip/hip_runtime.h>
#include <hip/hip_bf16.h>
#include <math.h>

typedef __attribute__((ext_vector_type(16))) _Float16 v16h;
typedef __attribute__((ext_vector_type(8)))  float    v8f;

#define LN_EPS  1e-5f
#define COS_EPS 1e-8f

// ---------------- helpers ----------------

__device__ __forceinline__ float waveReduceSum(float v) {
#pragma unroll
  for (int o = 16; o > 0; o >>= 1) v += __shfl_xor(v, o, 32);
  return v;
}

// float atomic max via sign-aware integer atomics (correct for mixed signs,
// location initialized to -inf).
__device__ __forceinline__ void atomicMaxF(float* a, float v) {
  if (v >= 0.0f) atomicMax((int*)a, __float_as_int(v));
  else           atomicMin((unsigned int*)a, __float_as_uint(v));
}

// ---------------- kernels ----------------

// emb = concat(emb_user, emb_item); out(acc) = emb   (embs[0] term of the mean)
__global__ void k_init_emb(const float* __restrict__ eu, const float* __restrict__ ei,
                           long nu, long tot, float* __restrict__ emb, float* __restrict__ out) {
  long i = (long)blockIdx.x * blockDim.x + threadIdx.x;
  if (i >= tot) return;
  float v = (i < nu) ? eu[i] : ei[i - nu];
  emb[i] = v;
  out[i] = v;
}

// eigs = layer_norm(L_low + sqrt_beta^2 * L_high), one wave per 64-wide row
__global__ void k_eigs_ln(const float* __restrict__ llow, const float* __restrict__ lhigh,
                          const float* __restrict__ sb, float* __restrict__ y, int N) {
  int lane = threadIdx.x & 31, wave = threadIdx.x >> 5;
  int row = blockIdx.x * (blockDim.x >> 5) + wave;
  if (row >= N) return;
  size_t base = (size_t)row * 64;
  float2 a = ((const float2*)(llow + base))[lane];
  float2 b = ((const float2*)(lhigh + base))[lane];
  float2 s = ((const float2*)sb)[lane];
  float x0 = a.x + s.x * s.x * b.x;
  float x1 = a.y + s.y * s.y * b.y;
  float mu = waveReduceSum(x0 + x1) * (1.0f / 64.0f);
  float d0 = x0 - mu, d1 = x1 - mu;
  float var = waveReduceSum(d0 * d0 + d1 * d1) * (1.0f / 64.0f);
  float r = rsqrtf(var + LN_EPS);
  ((float2*)(y + base))[lane] = make_float2(d0 * r, d1 * r);
}

// y = layer_norm(x), one wave per row
__global__ void k_ln(const float* __restrict__ x, float* __restrict__ y, int N) {
  int lane = threadIdx.x & 31, wave = threadIdx.x >> 5;
  int row = blockIdx.x * (blockDim.x >> 5) + wave;
  if (row >= N) return;
  size_t base = (size_t)row * 64;
  float2 v = ((const float2*)(x + base))[lane];
  float mu = waveReduceSum(v.x + v.y) * (1.0f / 64.0f);
  float d0 = v.x - mu, d1 = v.y - mu;
  float var = waveReduceSum(d0 * d0 + d1 * d1) * (1.0f / 64.0f);
  float r = rsqrtf(var + LN_EPS);
  ((float2*)(y + base))[lane] = make_float2(d0 * r, d1 * r);
}

// Edge-batch dot products via WMMA: 16 edges per wave, two chained
// v_wmma_f32_16x16x32_f16 (K=0..31, K=32..63); diagonal of the 16x16 product
// is the per-edge dot. out[e] = alpha*dot (+ exp(*lamp)*addv[e] if addv).
__global__ void k_edge_dot(const float* __restrict__ feat,
                           const int* __restrict__ src, const int* __restrict__ dst,
                           int E, float alpha,
                           const float* __restrict__ addv, const float* __restrict__ lamp,
                           float* __restrict__ out) {
  int lane = threadIdx.x & 31;
  int wave = threadIdx.x >> 5;
  long ebase = ((long)blockIdx.x * (blockDim.x >> 5) + wave) * 16;
  if (ebase >= E) return;                 // wave-uniform exit; EXEC all-1 for WMMA
  int l16 = lane & 15, hi = lane >> 4;

  long eL = ebase + l16; if (eL > E - 1) eL = E - 1;   // clamp loads at tail
  const float* pa = feat + (size_t)src[eL] * 64;       // A row for edge eL
  const float* pb = feat + (size_t)dst[eL] * 64;       // B col for edge eL

  // A (16x32 f16): lane hi=0 holds K {0..15}, hi=1 holds K {16..31}
  v16h a0, a1, b0, b1;
#pragma unroll
  for (int j = 0; j < 8; ++j) {
    a0[j]     = (_Float16)pa[16 * hi + j];
    a0[j + 8] = (_Float16)pa[16 * hi + 8 + j];
    a1[j]     = (_Float16)pa[32 + 16 * hi + j];
    a1[j + 8] = (_Float16)pa[32 + 16 * hi + 8 + j];
  }
  // B (32x16 f16): lane holds column, K = 16*hi + j (contiguous 16 floats)
#pragma unroll
  for (int j = 0; j < 16; ++j) {
    b0[j] = (_Float16)pb[16 * hi + j];
    b1[j] = (_Float16)pb[32 + 16 * hi + j];
  }

  v8f c = {};
  c = __builtin_amdgcn_wmma_f32_16x16x32_f16(false, a0, false, b0, (short)0, c, false, false);
  c = __builtin_amdgcn_wmma_f32_16x16x32_f16(false, a1, false, b1, (short)0, c, false, false);

  // Diagonal extraction per 32-bit C/D layout:
  // lanes 0..7 hold D[i][i] i=lane in c[lane]; lanes 24..31 hold i=lane-16 in c[lane-24]
  int idx = (lane < 8) ? lane : (lane - 24);
  float dsel = 0.0f;
#pragma unroll
  for (int j = 0; j < 8; ++j) if (idx == j) dsel = c[j];

  long eW = -1;
  if (lane < 8)        eW = ebase + lane;
  else if (lane >= 24) eW = ebase + (lane - 16);
  if (eW >= 0 && eW < E) {
    float v = alpha * dsel;
    if (addv) v += expf(lamp[0]) * addv[eW];
    out[eW] = v;
  }
}

// me[n] = motif_embs[l][motif_ids[n]]
__global__ void k_me(const float* __restrict__ membs, const int* __restrict__ mids,
                     float* __restrict__ me, int N) {
  int i = blockIdx.x * blockDim.x + threadIdx.x;
  if (i < N) me[i] = membs[mids[i]];
}

// per-block partial sums of (ei*ej, ei^2, ej^2) over edges
__global__ void k_motif_reduce(const float* __restrict__ me, const int* __restrict__ src,
                               const int* __restrict__ dst, int E, float* __restrict__ red) {
  float pij = 0.f, pii = 0.f, pjj = 0.f;
  for (long e = (long)blockIdx.x * blockDim.x + threadIdx.x; e < E;
       e += (long)gridDim.x * blockDim.x) {
    float a = me[src[e]], b = me[dst[e]];
    pij = fmaf(a, b, pij); pii = fmaf(a, a, pii); pjj = fmaf(b, b, pjj);
  }
  pij = waveReduceSum(pij); pii = waveReduceSum(pii); pjj = waveReduceSum(pjj);
  __shared__ float sm[3][8];
  int lane = threadIdx.x & 31, w = threadIdx.x >> 5;
  if (lane == 0) { sm[0][w] = pij; sm[1][w] = pii; sm[2][w] = pjj; }
  __syncthreads();
  if (threadIdx.x == 0) {
    float A = 0.f, B = 0.f, C = 0.f;
    for (int i = 0; i < (int)(blockDim.x >> 5); ++i) { A += sm[0][i]; B += sm[1][i]; C += sm[2][i]; }
    red[blockIdx.x * 3 + 0] = A; red[blockIdx.x * 3 + 1] = B; red[blockIdx.x * 3 + 2] = C;
  }
}

// reduce partials, c = gammas[l] * num / max(||ei||*||ej||, eps)
__global__ void k_motif_final(const float* __restrict__ red, int nb,
                              const float* __restrict__ gammas, int l,
                              float* __restrict__ cptr) {
  float a = 0.f, b = 0.f, c = 0.f;
  for (int i = threadIdx.x; i < nb; i += blockDim.x) {
    a += red[3 * i]; b += red[3 * i + 1]; c += red[3 * i + 2];
  }
  a = waveReduceSum(a); b = waveReduceSum(b); c = waveReduceSum(c);
  __shared__ float sm[3][8];
  int lane = threadIdx.x & 31, w = threadIdx.x >> 5;
  if (lane == 0) { sm[0][w] = a; sm[1][w] = b; sm[2][w] = c; }
  __syncthreads();
  if (threadIdx.x == 0) {
    float A = 0.f, B = 0.f, C = 0.f;
    for (int i = 0; i < (int)(blockDim.x >> 5); ++i) { A += sm[0][i]; B += sm[1][i]; C += sm[2][i]; }
    float den = fmaxf(sqrtf(B) * sqrtf(C), COS_EPS);
    cptr[0] = gammas[l] * (A / den);
  }
}

// mz = [m0(N), z0(N), m1(N), z1(N)]; m -> -inf, z -> 0
__global__ void k_init_mz(float* __restrict__ mz, int N) {
  long i = (long)blockIdx.x * blockDim.x + threadIdx.x;
  if (i >= 4L * N) return;
  long blk = i / N;
  mz[i] = (blk == 0 || blk == 2) ? -INFINITY : 0.0f;
}

__global__ void k_smax(const int* __restrict__ src, const float* __restrict__ s0,
                       const float* __restrict__ mv, const float* __restrict__ cptr,
                       float* __restrict__ m0, float* __restrict__ m1, int E) {
  long e = (long)blockIdx.x * blockDim.x + threadIdx.x;
  if (e >= E) return;
  int s = src[e];
  atomicMaxF(m0 + s, s0[e]);
  atomicMaxF(m1 + s, cptr[0] * mv[e]);
}

__global__ void k_ssum(const int* __restrict__ src, const float* __restrict__ s0,
                       const float* __restrict__ mv, const float* __restrict__ cptr,
                       const float* __restrict__ m0, float* __restrict__ z0,
                       const float* __restrict__ m1, float* __restrict__ z1, int E) {
  long e = (long)blockIdx.x * blockDim.x + threadIdx.x;
  if (e >= E) return;
  int s = src[e];
  atomicAdd(z0 + s, expf(s0[e] - m0[s]));
  atomicAdd(z1 + s, expf(cptr[0] * mv[e] - m1[s]));
}

__global__ void k_zero(float* __restrict__ p, long n) {
  long i = (long)blockIdx.x * blockDim.x + threadIdx.x;
  if (i < n) p[i] = 0.0f;
}

// emb[src] += 0.5*(softmax0 + softmax1) * y[dst]; 16 threads per edge, 4 dims each
__global__ void k_agg(const float* __restrict__ y, const int* __restrict__ src,
                      const int* __restrict__ dst, const float* __restrict__ s0,
                      const float* __restrict__ mv, const float* __restrict__ cptr,
                      const float* __restrict__ m0, const float* __restrict__ z0,
                      const float* __restrict__ m1, const float* __restrict__ z1,
                      int E, float* __restrict__ emb) {
  long t = (long)blockIdx.x * blockDim.x + threadIdx.x;
  if (t >= (long)E * 16) return;
  long e = t >> 4;
  int part = (int)(t & 15);
  int s = src[e], dv = dst[e];
  float a0 = expf(s0[e] - m0[s]) / z0[s];
  float s1 = cptr[0] * mv[e];
  float a1 = expf(s1 - m1[s]) / z1[s];
  float a = 0.5f * (a0 + a1);
  const float* py = y + (size_t)dv * 64 + part * 4;
  float* pe = emb + (size_t)s * 64 + part * 4;
#pragma unroll
  for (int j = 0; j < 4; ++j) atomicAdd(pe + j, a * py[j]);
}

__global__ void k_addacc(float* __restrict__ out, const float* __restrict__ emb, long n) {
  long i = (long)blockIdx.x * blockDim.x + threadIdx.x;
  if (i < n) out[i] += emb[i];
}

__global__ void k_scale(float* __restrict__ out, long n, float sc) {
  long i = (long)blockIdx.x * blockDim.x + threadIdx.x;
  if (i < n) out[i] *= sc;
}

// ---------------- launch ----------------

extern "C" void kernel_launch(void* const* d_in, const int* in_sizes, int n_in,
                              void* d_out, int out_size, void* d_ws, size_t ws_size,
                              hipStream_t stream) {
  const float* emb_user   = (const float*)d_in[0];
  const float* emb_item   = (const float*)d_in[1];
  const float* sqrt_beta  = (const float*)d_in[2];
  const float* L_low      = (const float*)d_in[3];
  const float* L_high     = (const float*)d_in[4];
  const float* lambda0s   = (const float*)d_in[5];
  const float* gammas     = (const float*)d_in[6];
  const float* motif_embs = (const float*)d_in[7];
  const float* motif_vals = (const float*)d_in[8];
  const int*   edge_index = (const int*)d_in[9];
  const int*   motif_ids  = (const int*)d_in[10];
  float* out = (float*)d_out;

  const int D = 64;
  int Nu = in_sizes[0] / D, Ni = in_sizes[1] / D;
  int N = Nu + Ni;
  int E = in_sizes[8];
  int n_layers = in_sizes[5];
  int num_motifs = (n_layers > 0) ? in_sizes[7] / n_layers : 0;
  const int* src = edge_index;
  const int* dst = edge_index + E;

  size_t NF = (size_t)N * D;
  const int NB_RED = 1024;
  float* W = (float*)d_ws;
  float* f_y   = W;  W += NF;                  // eigs, then per-layer y
  float* f_emb = W;  W += NF;                  // current embedding
  float* f_ye  = W;  W += (size_t)E;           // eig edge dots (reused all layers)
  float* f_s0  = W;  W += (size_t)E;           // edge scores s0
  float* f_mz  = W;  W += 4 * (size_t)N;       // m0,z0,m1,z1
  float* f_me  = W;  W += (size_t)N;           // per-node motif scalar
  float* f_red = W;  W += 3 * (size_t)NB_RED;  // reduction partials
  float* f_c   = W;  W += 4;                   // gamma*sim scalar
  float* m0 = f_mz, *z0 = f_mz + N, *m1 = f_mz + 2 * (size_t)N, *z1 = f_mz + 3 * (size_t)N;

  const int B = 256;
  unsigned gNF = (unsigned)((NF + B - 1) / B);
  unsigned gE  = (unsigned)(((long)E + B - 1) / B);
  unsigned gLN = (unsigned)((N + 7) / 8);                 // 8 waves/block, wave per row
  unsigned gDOT = (unsigned)((((long)E + 15) / 16 + 7) / 8); // 16 edges/wave, 8 waves/block
  unsigned gN  = (unsigned)((N + B - 1) / B);
  unsigned gMZ = (unsigned)((4L * N + B - 1) / B);
  long tasks = (long)E * 16;
  unsigned gAGG = (unsigned)((tasks + B - 1) / B);
  float inv_sqrt_d = 1.0f / sqrtf((float)D);

  // embs[0] term + current embedding
  k_init_emb<<<gNF, B, 0, stream>>>(emb_user, emb_item, (long)Nu * D, (long)NF, f_emb, out);
  // eigs (into f_y), then edge eig-dots once (eigs are layer-invariant)
  k_eigs_ln<<<gLN, B, 0, stream>>>(L_low, L_high, sqrt_beta, f_y, N);
  k_edge_dot<<<gDOT, B, 0, stream>>>(f_y, src, dst, E, 1.0f, nullptr, nullptr, f_ye);

  for (int l = 0; l < n_layers; ++l) {
    k_ln<<<gLN, B, 0, stream>>>(f_emb, f_y, N);   // f_y now holds y (eigs no longer needed)
    k_me<<<gN, B, 0, stream>>>(motif_embs + (size_t)l * num_motifs, motif_ids, f_me, N);
    k_motif_reduce<<<NB_RED, B, 0, stream>>>(f_me, src, dst, E, f_red);
    k_motif_final<<<1, B, 0, stream>>>(f_red, NB_RED, gammas, l, f_c);
    // s0 = dot(y[src],y[dst])/sqrt(D) + exp(lambda0)*ye   (WMMA path)
    k_edge_dot<<<gDOT, B, 0, stream>>>(f_y, src, dst, E, inv_sqrt_d, f_ye, lambda0s + l, f_s0);
    // two segment softmaxes + aggregation
    k_init_mz<<<gMZ, B, 0, stream>>>(f_mz, N);
    k_smax<<<gE, B, 0, stream>>>(src, f_s0, motif_vals, f_c, m0, m1, E);
    k_ssum<<<gE, B, 0, stream>>>(src, f_s0, motif_vals, f_c, m0, z0, m1, z1, E);
    k_zero<<<gNF, B, 0, stream>>>(f_emb, (long)NF);
    k_agg<<<gAGG, B, 0, stream>>>(f_y, src, dst, f_s0, motif_vals, f_c, m0, z0, m1, z1, E, f_emb);
    k_addacc<<<gNF, B, 0, stream>>>(out, f_emb, (long)NF);
  }
  k_scale<<<gNF, B, 0, stream>>>(out, (long)NF, 1.0f / (float)(n_layers + 1));
}
